// Extracter_31490700215149
// MI455X (gfx1250) — matrix operand order repaired and checked by reading, not compiled, not run on hardware
//
#include <hip/hip_runtime.h>
#include <hip/hip_bf16.h>

// ---------------------------------------------------------------------------
// Problem constants (from reference):
//   N=2 batches, C3=256, grid 48x48 -> L=2304 columns, D=256*9=2304 dims.
//   GEMM: R[n] = Kn[n] (2304xD) x Qn[n] (Dx2304) -> 2304x2304 cosine sims.
//   top-3 over rows per query; then 9-tap shifted gathers for T3/T2/T1.
// Workspace layout (bytes):
//   Qn  f16 : 2*2304*2304*2 = 21,233,664
//   Kn  f16 : 21,233,664
//   R   f32 : 2*2304*2304*4 = 42,467,328
//   idx i32 : 3*2*2304*4    = 55,296        (total ~85 MB)
// ---------------------------------------------------------------------------

typedef _Float16 v16h __attribute__((ext_vector_type(16)));
typedef _Float16 v8h  __attribute__((ext_vector_type(8)));
typedef float    v8f  __attribute__((ext_vector_type(8)));

#define LQ 2304
#define DD 2304
#define GRID 48

// ---------------------------------------------------------------------------
// Kernel A: unfold (3x3, stride 1, pad 1) + L2 normalize -> f16 feature rows.
// One block (256 threads) per (matrix, n, column). Thread t owns channel t,
// produces its 9 patch values, participates in sum-of-squares reduction, then
// the block cooperatively streams 2304 halves out as 16B chunks.
// ---------------------------------------------------------------------------
__global__ __launch_bounds__(256)
void feat_norm_kernel(const float* __restrict__ lr,
                      const float* __restrict__ refsr,
                      _Float16* __restrict__ Qn,
                      _Float16* __restrict__ Kn) {
  __shared__ float red[256];
  __shared__ __align__(16) _Float16 hbuf[DD];

  int b   = blockIdx.x;          // [0, 2*2*2304)
  int mat = b / (2 * LQ);
  int rem = b % (2 * LQ);
  int n   = rem / LQ;
  int l   = rem % LQ;
  int qy  = l / GRID, qx = l % GRID;

  const float* src = (mat ? refsr : lr) + (size_t)n * 256 * GRID * GRID;
  _Float16*    dst = (mat ? Kn : Qn) + ((size_t)(n * LQ + l)) * DD;

  int c = threadIdx.x;
  const float* sc = src + (size_t)c * (GRID * GRID);

  float vals[9];
  float ss = 0.f;
#pragma unroll
  for (int i = 0; i < 3; ++i) {
#pragma unroll
    for (int j = 0; j < 3; ++j) {
      int yy = qy - 1 + i, xx = qx - 1 + j;
      float v = (yy >= 0 && yy < GRID && xx >= 0 && xx < GRID)
                    ? sc[yy * GRID + xx] : 0.f;
      vals[i * 3 + j] = v;
      ss += v * v;
    }
  }
  red[c] = ss;
  __syncthreads();
#pragma unroll
  for (int s = 128; s > 0; s >>= 1) {
    if (c < s) red[c] += red[c + s];
    __syncthreads();
  }
  float scale = 1.f / fmaxf(sqrtf(red[0]), 1e-12f);

#pragma unroll
  for (int u = 0; u < 9; ++u)
    hbuf[c * 9 + u] = (_Float16)(vals[u] * scale);
  __syncthreads();

  const uint4* s4 = (const uint4*)hbuf;
  uint4*       d4 = (uint4*)dst;            // row base is 4608B -> 16B aligned
  for (int i = c; i < (DD * 2) / 16; i += 256) d4[i] = s4[i];
}

// ---------------------------------------------------------------------------
// Kernel B: WMMA GEMM  R[n][q][r] = sum_d Kn[n][r][d] * Qn[n][q][d]
// One wave per 32x64 output tile: 2 A-frags x 4 B-frags -> 8 WMMAs per
// k-step (1.33 wmma per fragment load vs 1.0 for 32x32), ~21 FLOP per L2
// byte. Fragment layouts used here (contiguous k, not the ISA table order):
//   A (16x32 f16): lane M = l%16; halves[i] = K[(l>>4)*16 + i]  (contiguous,
//                  matching B).
//   B (32x16 f16): lane N = l%16; halves[i] = K[(l>>4)*16 + i]  (contiguous!)
// Result transposed through padded LDS so R is stored query-major.
// 4 waves per block keeps the padded transpose buffer at 33 KB LDS.
// ---------------------------------------------------------------------------
__device__ __forceinline__
v16h load_afrag(const _Float16* __restrict__ base, int row, int kk, int lane) {
  const _Float16* p = base + (size_t)row * DD + kk * 32 + ((lane >> 4) << 4);
  v8h lo = *(const v8h*)p;
  v8h hi = *(const v8h*)(p + 8);
  v16h a;
#pragma unroll
  for (int i = 0; i < 8; ++i) { a[i] = lo[i]; a[i + 8] = hi[i]; }
  return a;
}

__device__ __forceinline__
v16h load_bfrag(const _Float16* __restrict__ base, int q, int kk, int lane) {
  const _Float16* p = base + (size_t)q * DD + kk * 32 + ((lane >> 4) << 4);
  return *(const v16h*)p;                 // 32B aligned -> 2x b128
}

__global__ __launch_bounds__(128)
void simgemm_kernel(const _Float16* __restrict__ Kn,
                    const _Float16* __restrict__ Qn,
                    float* __restrict__ R) {
  __shared__ float lds[4 * 32 * 65];      // per-wave 32x64 tile, padded stride

  int lane = threadIdx.x & 31;
  int wid  = threadIdx.x >> 5;
  int tile = blockIdx.x * 4 + wid;        // 5184 tiles = 1296 blocks * 4
  int n  = tile / (72 * 36);
  int t2 = tile % (72 * 36);
  int rt = t2 / 36, qt = t2 % 36;         // 72 r-tiles of 32, 36 q-tiles of 64

  const _Float16* Kb = Kn + (size_t)n * LQ * DD;
  const _Float16* Qb = Qn + (size_t)n * LQ * DD;

  int r0 = rt * 32 + (lane & 15), r1 = r0 + 16;
  int qbase = qt * 64 + (lane & 15);

  v8f acc[2][4] = {};
  for (int kk = 0; kk < DD / 32; ++kk) {
    v16h a0 = load_afrag(Kb, r0, kk, lane);
    v16h a1 = load_afrag(Kb, r1, kk, lane);
    v16h b[4];
#pragma unroll
    for (int j = 0; j < 4; ++j) b[j] = load_bfrag(Qb, qbase + 16 * j, kk, lane);
#pragma unroll
    for (int j = 0; j < 4; ++j) {
      acc[0][j] = __builtin_amdgcn_wmma_f32_16x16x32_f16(
          false, a0, false, b[j], (short)0, acc[0][j], false, false);
      acc[1][j] = __builtin_amdgcn_wmma_f32_16x16x32_f16(
          false, a1, false, b[j], (short)0, acc[1][j], false, false);
    }
  }

  // C/D layout: element (M,N) = (v + 8*(lane>>4), lane&15). Write [r][q] tile.
  float* L = lds + wid * (32 * 65);
  int mrow = 8 * (lane >> 4);
  int ncol = lane & 15;
#pragma unroll
  for (int bi = 0; bi < 2; ++bi)
#pragma unroll
    for (int bj = 0; bj < 4; ++bj)
#pragma unroll
      for (int v = 0; v < 8; ++v)
        L[(bi * 16 + v + mrow) * 65 + bj * 16 + ncol] = acc[bi][bj][v];
  __syncthreads();

  // Transposed, coalesced store: lane owns 2 query rows, 128B contiguous each.
#pragma unroll
  for (int h = 0; h < 2; ++h) {
    int ql = lane + 32 * h;
    float* dst = R + ((size_t)(n * LQ + qt * 64 + ql)) * LQ + rt * 32;
#pragma unroll
    for (int i = 0; i < 8; ++i) {
      float4 o = { L[(4 * i + 0) * 65 + ql], L[(4 * i + 1) * 65 + ql],
                   L[(4 * i + 2) * 65 + ql], L[(4 * i + 3) * 65 + ql] };
      *(float4*)(dst + 4 * i) = o;
    }
  }
}

// ---------------------------------------------------------------------------
// Kernel C: per-query top-3 over 2304 contiguous sims. One wave per query;
// per-lane running top-3 then log2(32) shfl_xor merges. Writes S output and
// hard indices for the transfer stage.
// ---------------------------------------------------------------------------
__device__ __forceinline__
void top3_insert(float v, int r, float& v0, float& v1, float& v2,
                 int& i0, int& i1, int& i2) {
  if (v > v2) {
    if (v > v1) {
      v2 = v1; i2 = i1;
      if (v > v0) { v1 = v0; i1 = i0; v0 = v; i0 = r; }
      else        { v1 = v;  i1 = r; }
    } else { v2 = v; i2 = r; }
  }
}

__global__ __launch_bounds__(256)
void topk_kernel(const float* __restrict__ R,
                 float* __restrict__ Sout,
                 int* __restrict__ idxWS) {
  int lane = threadIdx.x & 31;
  int wid  = threadIdx.x >> 5;
  int q = blockIdx.x * 8 + wid;           // [0, 4608)
  const float* row = R + (size_t)q * LQ;

  float v0 = -1e30f, v1 = -1e30f, v2 = -1e30f;
  int   i0 = 0, i1 = 0, i2 = 0;
  for (int it = 0; it < LQ / 32; ++it) {
    int r = it * 32 + lane;
    top3_insert(row[r], r, v0, v1, v2, i0, i1, i2);
  }
  for (int off = 16; off; off >>= 1) {
    float w0 = __shfl_xor(v0, off), w1 = __shfl_xor(v1, off),
          w2 = __shfl_xor(v2, off);
    int   j0 = __shfl_xor(i0, off), j1 = __shfl_xor(i1, off),
          j2 = __shfl_xor(i2, off);
    top3_insert(w0, j0, v0, v1, v2, i0, i1, i2);
    top3_insert(w1, j1, v0, v1, v2, i0, i1, i2);
    top3_insert(w2, j2, v0, v1, v2, i0, i1, i2);
  }
  if (lane == 0) {
    int n = q / LQ, lq = q % LQ;
    Sout[0 * (2 * LQ) + n * LQ + lq] = v0;   // S: [3][2][1][48][48]
    Sout[1 * (2 * LQ) + n * LQ + lq] = v1;
    Sout[2 * (2 * LQ) + n * LQ + lq] = v2;
    idxWS[(0 * 2 + n) * LQ + lq] = i0;
    idxWS[(1 * 2 + n) * LQ + lq] = i1;
    idxWS[(2 * 2 + n) * LQ + lq] = i2;
  }
}

// ---------------------------------------------------------------------------
// Kernel D: fold(gather(unfold(ref))) / div collapses to a 9-tap shifted
// gather. For k=3s, stride=s, pad=s: output (y,x) sums ref pixels at
// (y + s*(ry-gy), x + s*(rx-gx)) over gy,gx in {g0-1,g0,g0+1} clipped to
// [0,48), g0 = floor(y/s); divisor is just the in-range cell count ny*nx.
// ---------------------------------------------------------------------------
template <int SC, int C>
__global__ __launch_bounds__(256)
void transfer_kernel(const float* __restrict__ ref,
                     const int* __restrict__ idxWS,
                     float* __restrict__ out, int total) {
  int t = blockIdx.x * 256 + threadIdx.x;
  if (t >= total) return;
  const int H = GRID * SC, W = GRID * SC;

  int x  = t % W;
  int y  = (t / W) % H;
  int c  = (t / (W * H)) % C;
  int n  = (t / (W * H * C)) % 2;
  int k  = t / (W * H * C * 2);

  int g0y = y / SC, g0x = x / SC;
  int gy0 = max(g0y - 1, 0), gy1 = min(g0y + 1, GRID - 1);
  int gx0 = max(g0x - 1, 0), gx1 = min(g0x + 1, GRID - 1);

  const int*   idx = idxWS + ((size_t)(k * 2 + n)) * LQ;
  const float* rc  = ref + ((size_t)(n * C + c)) * H * W;

  float sum = 0.f;
  for (int gy = gy0; gy <= gy1; ++gy) {
    for (int gx = gx0; gx <= gx1; ++gx) {
      int r  = idx[gy * GRID + gx];
      int ry = r / GRID, rx = r % GRID;
      int sy = y + SC * (ry - gy);
      int sx = x + SC * (rx - gx);
      if (sy >= 0 && sy < H && sx >= 0 && sx < W) sum += rc[sy * W + sx];
    }
  }
  float inv = 1.f / (float)((gy1 - gy0 + 1) * (gx1 - gx0 + 1));
  out[t] = sum * inv;
}

// ---------------------------------------------------------------------------
extern "C" void kernel_launch(void* const* d_in, const int* in_sizes, int n_in,
                              void* d_out, int out_size, void* d_ws,
                              size_t ws_size, hipStream_t stream) {
  const float* lr    = (const float*)d_in[0];
  const float* refsr = (const float*)d_in[1];
  const float* ref1  = (const float*)d_in[2];
  const float* ref2  = (const float*)d_in[3];
  const float* ref3  = (const float*)d_in[4];
  float* out = (float*)d_out;

  char* ws = (char*)d_ws;
  _Float16* Qn = (_Float16*)ws;
  _Float16* Kn = Qn + (size_t)2 * LQ * DD;
  float*    R  = (float*)(Kn + (size_t)2 * LQ * DD);
  int*   idxWS = (int*)(R + (size_t)2 * LQ * LQ);

  // A: features (2 matrices x 2 batches x 2304 columns)
  feat_norm_kernel<<<2 * 2 * LQ, 256, 0, stream>>>(lr, refsr, Qn, Kn);

  // B: similarity GEMM (5184 wave-tiles of 32x64 / 4 waves per block)
  simgemm_kernel<<<(2 * 72 * 36) / 4, 128, 0, stream>>>(Kn, Qn, R);

  // C: top-3 + S output + hard indices (4608 queries / 8 waves per block)
  topk_kernel<<<(2 * LQ) / 8, 256, 0, stream>>>(R, out, idxWS);

  // D: transfer gathers. Output offsets: S=13824, then T3, T2, T1.
  const int t3 = 3 * 2 * 256 * 48 * 48;    //  3,538,944
  const int t2 = 3 * 2 * 128 * 96 * 96;    //  7,077,888
  const int t1 = 3 * 2 * 64 * 192 * 192;   // 14,155,776
  float* o3 = out + 13824;
  float* o2 = o3 + t3;
  float* o1 = o2 + t2;
  transfer_kernel<1, 256><<<(t3 + 255) / 256, 256, 0, stream>>>(ref3, idxWS, o3, t3);
  transfer_kernel<2, 128><<<(t2 + 255) / 256, 256, 0, stream>>>(ref2, idxWS, o2, t2);
  transfer_kernel<4, 64><<<(t1 + 255) / 256, 256, 0, stream>>>(ref1, idxWS, o1, t1);
}
